// GATTransformerNet_88304527606163
// MI455X (gfx1250) — matrix operand order repaired and checked by reading, not compiled, not run on hardware
//
#include <hip/hip_runtime.h>
#include <hip/hip_bf16.h>
#include <math.h>
#include <stdint.h>

#define N_NODES 20000
#define N_EDGES 160000
#define IN_CH   128
#define HEADS   16
#define CDIM    32
#define HCH     512
#define ED_RAW  93
#define ED_PAD  96
#define NGRAPH  64

typedef __attribute__((ext_vector_type(16))) _Float16 v16h;
typedef __attribute__((ext_vector_type(8)))  _Float16 h8;
typedef __attribute__((ext_vector_type(8)))  float    v8f;

// ---------------- conversion / packing ----------------

__global__ void k_cvt_f16(const float* __restrict__ in, _Float16* __restrict__ out, long n) {
  long i = (long)blockIdx.x * blockDim.x + threadIdx.x;
  if (i < n) out[i] = (_Float16)in[i];
}

// edge_attr [E x 93] f32 -> [E x 96] f16, zero padded
__global__ void k_cvt_pad_edge(const float* __restrict__ ea, _Float16* __restrict__ out) {
  long i = (long)blockIdx.x * blockDim.x + threadIdx.x;
  long total = (long)N_EDGES * ED_PAD;
  if (i >= total) return;
  int d = (int)(i % ED_PAD);
  long e = i / ED_PAD;
  out[i] = (_Float16)(d < ED_RAW ? ea[e * ED_RAW + d] : 0.f);
}

// Pack W[Ksrc x 512] (f32, row-major) into WMMA B-fragment order (f16).
// GEMM read: lane on column-tile t, k-step s loads 16 contiguous halves at
// Wp[((t*Ks + s)*32 + lane)*16]; element i <-> K = s*32 + (lane>=16?16:0) + i,
// column N = t*16 + (lane&15).   (contiguous k order, same as the A fragment)
__global__ void k_pack_w(const float* __restrict__ W, _Float16* __restrict__ Wp,
                         int Kpad, int Ksrc) {
  int Ks = Kpad >> 5;
  int total = 32 * Ks * 32;                 // colTiles * ksteps * lanes
  int idx = blockIdx.x * blockDim.x + threadIdx.x;
  if (idx >= total) return;
  int lane = idx & 31;
  int s    = (idx >> 5) % Ks;
  int t    = idx / (32 * Ks);
  int n    = t * 16 + (lane & 15);
  int kb   = s * 32 + ((lane >= 16) ? 16 : 0);
  _Float16* dst = Wp + (long)idx * 16;
#pragma unroll
  for (int i = 0; i < 16; ++i) {
    int kk = kb + i;
    dst[i] = (_Float16)((kk < Ksrc) ? W[(long)kk * HCH + n] : 0.f);
  }
}

// -------- WMMA GEMM: C[M x 512] = A[M x K] @ W (+bias), 4 col-tiles/wave ------
// grid (2, M/16), block 128 (4 waves). Wave w owns col tiles t0..t0+3 where
// t0 = blockIdx.x*16 + w*4, reusing one A fragment for 4 WMMAs per K-step.
__global__ void k_gemm_wmma(const _Float16* __restrict__ A,
                            const _Float16* __restrict__ Wp,
                            const float* __restrict__ bias,
                            float* __restrict__ C, int M, int K) {
  int lane    = threadIdx.x & 31;
  int wave    = threadIdx.x >> 5;
  int t0      = blockIdx.x * 16 + wave * 4;
  int rowBase = blockIdx.y * 16;
  int m  = lane & 15;
  int hi = (lane >= 16) ? 8 : 0;
  int Ks = K >> 5;
  const _Float16* arow = A  + (long)(rowBase + m) * K;
  const _Float16* bp   = Wp + ((long)t0 * Ks * 32 + lane) * 16;
  const long tileStride = (long)Ks * 512;   // halves per packed col-tile
  v8f acc[4] = {};
  for (int s = 0; s < Ks; ++s) {
    int k0 = s << 5;
    // A fragment, contiguous k order (not the ISA table order):
    //   lane<16: elems0-15 K=k0..k0+15; lane>=16: K=k0+16..k0+31 (matches packed B).
    h8 lo = *(const h8*)(arow + k0 + 2 * hi);
    h8 hh = *(const h8*)(arow + k0 + 2 * hi + 8);
    v16h a;
#pragma unroll
    for (int i = 0; i < 8; ++i) { a[i] = lo[i]; a[8 + i] = hh[i]; }
    __builtin_prefetch(arow + k0 + 32, 0, 1);
    const _Float16* bs_ = bp + (long)s * 512;
#pragma unroll
    for (int j = 0; j < 4; ++j) {
      v16h b = *(const v16h*)(bs_ + (long)j * tileStride);
      acc[j] = __builtin_amdgcn_wmma_f32_16x16x32_f16(false, a, false, b,
                                                      (short)0, acc[j], false, false);
    }
  }
  // C/D layout: lane<16 -> N=lane, M=r; lane>=16 -> N=lane-16, M=8+r
  float* crow = C + (long)(rowBase + hi) * HCH;
#pragma unroll
  for (int j = 0; j < 4; ++j) {
    int col  = (t0 + j) * 16 + m;
    float bv = bias ? bias[col] : 0.f;
#pragma unroll
    for (int r = 0; r < 8; ++r) crow[(long)r * HCH + col] = acc[j][r] + bv;
  }
}

// ---------------- CSR (by destination) build ----------------

__global__ void k_zero_i32(int* __restrict__ p, int n) {
  int i = blockIdx.x * blockDim.x + threadIdx.x;
  if (i < n) p[i] = 0;
}

__global__ void k_copy_i32(const int* __restrict__ a, int* __restrict__ b, int n) {
  int i = blockIdx.x * blockDim.x + threadIdx.x;
  if (i < n) b[i] = a[i];
}

__global__ void k_deg_count(const int* __restrict__ dst, int* __restrict__ deg) {
  int e = blockIdx.x * blockDim.x + threadIdx.x;
  if (e < N_EDGES) atomicAdd(&deg[dst[e]], 1);
}

// single-block exclusive scan of deg[0..N) -> rowptr[0..N]
__global__ void k_scan_rowptr(const int* __restrict__ deg, int* __restrict__ rowptr) {
  __shared__ int buf[256];
  __shared__ int base;
  int t = threadIdx.x;
  if (t == 0) base = 0;
  __syncthreads();
  for (int start = 0; start < N_NODES; start += 256) {
    int i = start + t;
    int val = (i < N_NODES) ? deg[i] : 0;
    buf[t] = val;
    __syncthreads();
    for (int off = 1; off < 256; off <<= 1) {
      int add = (t >= off) ? buf[t - off] : 0;
      __syncthreads();
      buf[t] += add;
      __syncthreads();
    }
    int incl = buf[t];
    int b = base;
    if (i < N_NODES) rowptr[i] = b + incl - val;
    __syncthreads();
    if (t == 0) base = b + buf[255];
    __syncthreads();
  }
  if (t == 0) rowptr[N_NODES] = base;
}

__global__ void k_csr_fill(const int* __restrict__ dst, int* __restrict__ cursor,
                           int* __restrict__ eidx) {
  int e = blockIdx.x * blockDim.x + threadIdx.x;
  if (e < N_EDGES) {
    int pos = atomicAdd(&cursor[dst[e]], 1);
    eidx[pos] = e;
  }
}

// ---------------- fused attention (softmax + aggregate + skip + leaky) -------
// One block (256 thr = 8 waves) per destination node. Thread t owns features
// f0 = t (head t>>5 == wave w) and f1 = t+256 (head 8+w), so each wave's 32
// lanes cover exactly the 32 channels of heads w and 8+w -> __shfl_xor
// allreduce gives per-head dot products to every lane. Online (max,sum)
// softmax stats in registers; per-edge scores parked in aS[idx*16+h].
__global__ void k_attn_fused(const float* __restrict__ q, const float* __restrict__ k,
                             const float* __restrict__ v, const float* __restrict__ e1,
                             const int* __restrict__ rowptr, const int* __restrict__ eidx,
                             const int* __restrict__ src,
                             float* __restrict__ aS, float* __restrict__ xout) {
  int d    = blockIdx.x;
  int t    = threadIdx.x;
  int lane = t & 31;
  int w    = t >> 5;
  int lo   = rowptr[d], hiE = rowptr[d + 1];
  const float SCL = 0.17677669529663687f;  // 1/sqrt(32)
  float qa = q[(long)d * HCH + t];
  float qb = q[(long)d * HCH + t + 256];
  float m0 = -__builtin_inff(), s0 = 0.f;
  float m1 = -__builtin_inff(), s1 = 0.f;
  for (int idx = lo; idx < hiE; ++idx) {
    int e  = eidx[idx];
    long sb = (long)src[e] * HCH;
    long eb = (long)e * HCH;
    float p0 = qa * (k[sb + t] + e1[eb + t]);
    float p1 = qb * (k[sb + t + 256] + e1[eb + t + 256]);
#pragma unroll
    for (int off = 16; off > 0; off >>= 1) {
      p0 += __shfl_xor(p0, off, 32);
      p1 += __shfl_xor(p1, off, 32);
    }
    float a0 = p0 * SCL, a1 = p1 * SCL;
    float nm0 = fmaxf(m0, a0);
    s0 = s0 * __expf(m0 - nm0) + __expf(a0 - nm0); m0 = nm0;
    float nm1 = fmaxf(m1, a1);
    s1 = s1 * __expf(m1 - nm1) + __expf(a1 - nm1); m1 = nm1;
    if (lane == 0) {
      aS[(long)idx * HEADS + w]     = a0;
      aS[(long)idx * HEADS + 8 + w] = a1;
    }
  }
  float acc0 = 0.f, acc1 = 0.f;
  for (int idx = lo; idx < hiE; ++idx) {
    int e  = eidx[idx];
    long sb = (long)src[e] * HCH;
    long eb = (long)e * HCH;
    float w0 = __expf(aS[(long)idx * HEADS + w] - m0) / s0;
    float w1 = __expf(aS[(long)idx * HEADS + 8 + w] - m1) / s1;
    acc0 += w0 * (v[sb + t] + e1[eb + t]);
    acc1 += w1 * (v[sb + t + 256] + e1[eb + t + 256]);
  }
  long ob = (long)d * HCH;
  float r0 = xout[ob + t] + acc0;          // skip/root already in xout
  float r1 = xout[ob + t + 256] + acc1;
  xout[ob + t]       = r0 > 0.f ? r0 : 0.2f * r0;   // fused leaky-relu
  xout[ob + t + 256] = r1 > 0.f ? r1 : 0.2f * r1;
}

// ---------------- pooling + MLP ----------------
// batch[n] = (n*B)//N is sorted: graph g spans [ceil(gN/B), ceil((g+1)N/B))
__global__ void k_pool(const float* __restrict__ x, float* __restrict__ p, int accum) {
  int g  = blockIdx.x;
  int lo = (g * N_NODES + NGRAPH - 1) / NGRAPH;
  int hi = ((g + 1) * N_NODES + NGRAPH - 1) / NGRAPH;
  float inv = 1.f / (float)(hi - lo);
  for (int col = threadIdx.x; col < HCH; col += blockDim.x) {
    float s = 0.f, mx = -__builtin_inff();
    for (int n = lo; n < hi; ++n) {
      float vv = x[(long)n * HCH + col];
      s += vv; mx = fmaxf(mx, vv);
    }
    float* pg = p + (long)g * 2 * HCH;
    if (accum) { pg[col] += s * inv; pg[HCH + col] += mx; }
    else       { pg[col]  = s * inv; pg[HCH + col]  = mx; }
  }
}

__global__ void k_fc1(const float* __restrict__ p, const float* __restrict__ W,
                      const float* __restrict__ b, float* __restrict__ h) {
  int o = blockIdx.x * blockDim.x + threadIdx.x;
  if (o >= NGRAPH * 256) return;
  int g = o >> 8, oc = o & 255;
  const float* pr = p + (long)g * (2 * HCH);
  float acc = b[oc];
  for (int i = 0; i < 2 * HCH; ++i) acc += pr[i] * W[(long)i * 256 + oc];
  h[o] = acc > 0.f ? acc : 0.f;
}

__global__ void k_fc2(const float* __restrict__ h, const float* __restrict__ W,
                      const float* __restrict__ b, float* __restrict__ out) {
  int o = blockIdx.x * blockDim.x + threadIdx.x;
  if (o >= NGRAPH * CDIM) return;
  int g = o >> 5, oc = o & 31;
  const float* hr = h + (long)g * 256;
  float acc = b[oc];
  for (int i = 0; i < 256; ++i) acc += hr[i] * W[(long)i * CDIM + oc];
  out[o] = acc;
}

// ---------------- host orchestration ----------------

extern "C" void kernel_launch(void* const* d_in, const int* in_sizes, int n_in,
                              void* d_out, int out_size, void* d_ws, size_t ws_size,
                              hipStream_t stream) {
  const float* x         = (const float*)d_in[0];
  const float* edge_attr = (const float*)d_in[1];
  const int*   ei        = (const int*)d_in[2];
  const int*   srcI      = ei;
  const int*   dstI      = ei + N_EDGES;
  const float* t1_Wq = (const float*)d_in[4],  *t1_bq = (const float*)d_in[5];
  const float* t1_Wk = (const float*)d_in[6],  *t1_bk = (const float*)d_in[7];
  const float* t1_Wv = (const float*)d_in[8],  *t1_bv = (const float*)d_in[9];
  const float* t1_We = (const float*)d_in[10];
  const float* t1_Ws = (const float*)d_in[11], *t1_bs = (const float*)d_in[12];
  const float* t2_Wq = (const float*)d_in[13], *t2_bq = (const float*)d_in[14];
  const float* t2_Wk = (const float*)d_in[15], *t2_bk = (const float*)d_in[16];
  const float* t2_Wv = (const float*)d_in[17], *t2_bv = (const float*)d_in[18];
  const float* t2_We = (const float*)d_in[19];
  const float* t2_Ws = (const float*)d_in[20], *t2_bs = (const float*)d_in[21];
  const float* fc1_W = (const float*)d_in[22], *fc1_b = (const float*)d_in[23];
  const float* fc2_W = (const float*)d_in[24], *fc2_b = (const float*)d_in[25];

  char* w = (char*)d_ws;
  auto take = [&](size_t bytes) -> void* {
    void* r = (void*)w; w += (bytes + 255) & ~(size_t)255; return r;
  };
  _Float16* xh   = (_Float16*)take((size_t)N_NODES * IN_CH * 2);
  _Float16* x1h  = (_Float16*)take((size_t)N_NODES * HCH * 2);
  _Float16* eh   = (_Float16*)take((size_t)N_EDGES * ED_PAD * 2);
  _Float16* Wpq  = (_Float16*)take((size_t)HCH * HCH * 2);   // sized for max K=512
  _Float16* Wpk  = (_Float16*)take((size_t)HCH * HCH * 2);
  _Float16* Wpv  = (_Float16*)take((size_t)HCH * HCH * 2);
  _Float16* Wps  = (_Float16*)take((size_t)HCH * HCH * 2);
  _Float16* Wpe  = (_Float16*)take((size_t)ED_PAD * HCH * 2);
  float* qb    = (float*)take((size_t)N_NODES * HCH * 4);
  float* kb    = (float*)take((size_t)N_NODES * HCH * 4);
  float* vb    = (float*)take((size_t)N_NODES * HCH * 4);
  float* e1    = (float*)take((size_t)N_EDGES * HCH * 4);
  float* aS    = (float*)take((size_t)N_EDGES * HEADS * 4);
  float* x1    = (float*)take((size_t)N_NODES * HCH * 4);
  float* x2    = (float*)take((size_t)N_NODES * HCH * 4);
  float* pbuf  = (float*)take((size_t)NGRAPH * 2 * HCH * 4);
  float* hbuf  = (float*)take((size_t)NGRAPH * 256 * 4);
  int* deg     = (int*)take((size_t)N_NODES * 4);
  int* rowptr  = (int*)take((size_t)(N_NODES + 1) * 4);
  int* cursor  = (int*)take((size_t)N_NODES * 4);
  int* eidx    = (int*)take((size_t)N_EDGES * 4);

  const int T = 256;
  auto blks = [](long n, int t) { return (unsigned)((n + t - 1) / t); };

  // -------- CSR by destination (built once, reused by both layers) --------
  k_zero_i32 <<<blks(N_NODES, T), T, 0, stream>>>(deg, N_NODES);
  k_deg_count<<<blks(N_EDGES, T), T, 0, stream>>>(dstI, deg);
  k_scan_rowptr<<<1, 256, 0, stream>>>(deg, rowptr);
  k_copy_i32 <<<blks(N_NODES, T), T, 0, stream>>>(rowptr, cursor, N_NODES);
  k_csr_fill <<<blks(N_EDGES, T), T, 0, stream>>>(dstI, cursor, eidx);

  // -------- precision staging --------
  k_cvt_f16     <<<blks((long)N_NODES * IN_CH, T), T, 0, stream>>>(x, xh, (long)N_NODES * IN_CH);
  k_cvt_pad_edge<<<blks((long)N_EDGES * ED_PAD, T), T, 0, stream>>>(edge_attr, eh);

  // -------- one transformer layer --------
  auto run_layer = [&](const _Float16* Ain, int K,
                       const float* Wq, const float* bq, const float* Wk_, const float* bk_,
                       const float* Wv_, const float* bv_, const float* We_,
                       const float* Ws_, const float* bs_, float* xout) {
    int Ks = K >> 5;
    int pt = 32 * Ks * 32;
    k_pack_w<<<blks(pt, T), T, 0, stream>>>(Wq,  Wpq, K, K);
    k_pack_w<<<blks(pt, T), T, 0, stream>>>(Wk_, Wpk, K, K);
    k_pack_w<<<blks(pt, T), T, 0, stream>>>(Wv_, Wpv, K, K);
    k_pack_w<<<blks(pt, T), T, 0, stream>>>(Ws_, Wps, K, K);
    int pte = 32 * (ED_PAD >> 5) * 32;
    k_pack_w<<<blks(pte, T), T, 0, stream>>>(We_, Wpe, ED_PAD, ED_RAW);

    dim3 gn(2, N_NODES / 16), ge(2, N_EDGES / 16), blk(128);
    k_gemm_wmma<<<gn, blk, 0, stream>>>(Ain, Wpq, bq,  qb,   N_NODES, K);
    k_gemm_wmma<<<gn, blk, 0, stream>>>(Ain, Wpk, bk_, kb,   N_NODES, K);
    k_gemm_wmma<<<gn, blk, 0, stream>>>(Ain, Wpv, bv_, vb,   N_NODES, K);
    k_gemm_wmma<<<gn, blk, 0, stream>>>(Ain, Wps, bs_, xout, N_NODES, K);   // skip/root
    k_gemm_wmma<<<ge, blk, 0, stream>>>(eh,  Wpe, nullptr, e1, N_EDGES, ED_PAD);

    k_attn_fused<<<N_NODES, 256, 0, stream>>>(qb, kb, vb, e1, rowptr, eidx, srcI, aS, xout);
  };

  // layer 1 (in = 128)
  run_layer(xh, IN_CH, t1_Wq, t1_bq, t1_Wk, t1_bk, t1_Wv, t1_bv, t1_We, t1_Ws, t1_bs, x1);
  k_pool<<<NGRAPH, 256, 0, stream>>>(x1, pbuf, 0);
  k_cvt_f16<<<blks((long)N_NODES * HCH, T), T, 0, stream>>>(x1, x1h, (long)N_NODES * HCH);

  // layer 2 (in = 512)
  run_layer(x1h, HCH, t2_Wq, t2_bq, t2_Wk, t2_bk, t2_Wv, t2_bv, t2_We, t2_Ws, t2_bs, x2);
  k_pool<<<NGRAPH, 256, 0, stream>>>(x2, pbuf, 1);   // fuses h = x3 + x4

  // MLP head
  k_fc1<<<NGRAPH, 256, 0, stream>>>(pbuf, fc1_W, fc1_b, hbuf);
  k_fc2<<<blks((long)NGRAPH * CDIM, T), T, 0, stream>>>(hbuf, fc2_W, fc2_b, (float*)d_out);
}
